// ControllerGNN_22179211116932
// MI455X (gfx1250) — compile-verified, matrix operand order checked by reading, not run on hardware
//
#include <hip/hip_runtime.h>
#include <hip/hip_bf16.h>

#define NA 4096
#define NE 524288

typedef __attribute__((ext_vector_type(16))) _Float16 v16h;
typedef __attribute__((ext_vector_type(4)))  _Float16 v4h;
typedef __attribute__((ext_vector_type(8)))  float    v8f;

// ---- monotone float <-> uint transform so atomicMax(u32) == float max ----
__device__ __forceinline__ unsigned encf(float f) {
    unsigned u = __float_as_uint(f);
    return (u & 0x80000000u) ? ~u : (u | 0x80000000u);
}
__device__ __forceinline__ float decf(unsigned u) {
    return (u & 0x80000000u) ? __uint_as_float(u & 0x7FFFFFFFu) : __uint_as_float(~u);
}
#define ENC_NEGINF 0x007FFFFFu   // encf(-inf)

// K offset inside one 16x32 f16 WMMA fragment for (half h, element e)
__device__ __forceinline__ int wmma_k(int h, int e) {
    int g = e >> 1, j = e & 1;
    return ((g & 4) ? 16 : 0) + h * 8 + (g & 3) * 2 + j;
}
// inverse map: kk (0..31) -> (half hh, element e)
__device__ __forceinline__ void inv_kk(int kk, int& hh, int& e) {
    hh = (kk >> 3) & 1;
    int g = ((kk >> 4) & 1) * 4 + ((kk >> 1) & 3);
    e = 2 * g + (kk & 1);
}

// -------- weight convert: f32 [K,N] -> f16 fragment-major [ks][nt][lane][e]
__global__ void convert_frag_kernel(const float* __restrict__ src, _Float16* __restrict__ dst,
                                    int K, int Kpad, int N) {
    int idx = blockIdx.x * 256 + threadIdx.x;
    if (idx >= Kpad * N) return;
    int e    = idx & 15;
    int lane = (idx >> 4) & 31;
    int rest = idx >> 9;
    int nt16 = N >> 4;
    int nt = rest % nt16, ks = rest / nt16;
    int h = lane >> 4, col = lane & 15;
    int k = ks * 32 + wmma_k(h, e);
    dst[idx] = (k < K) ? (_Float16)src[k * N + nt * 16 + col] : (_Float16)0.f;
}

__global__ void init_agg_kernel(unsigned* __restrict__ agg) {
    agg[blockIdx.x * 256 + threadIdx.x] = ENC_NEGINF;
}

// ------------------------- fused per-edge phi MLP + max-aggregate ---------
// One wave32 per 16-edge tile. MLP: K0PAD -> 64 -> 64 -> 256, relu on hiddens.
// All LDS buffers are wave-private and kept in WMMA fragment order.
template <int LAYER>
__global__ __launch_bounds__(128) void edge_phi_kernel(
    const float* __restrict__ x4,          // LAYER==1 node feats [NA,4]
    const _Float16* __restrict__ h64,      // LAYER==2 node feats [NA,64] (f16)
    const float* __restrict__ ea,          // [NE,4]
    const int* __restrict__ srcI, const int* __restrict__ dstI,
    const _Float16* __restrict__ W0, const float* __restrict__ b0,   // frag [K0PAD/32][4][32][16]
    const _Float16* __restrict__ W1, const float* __restrict__ b1,   // frag [2][4][32][16]
    const _Float16* __restrict__ W2, const float* __restrict__ b2,   // frag [2][16][32][16]
    unsigned* __restrict__ agg)            // [NA,256] encoded max
{
    constexpr int K0PAD = (LAYER == 1) ? 32 : 160;
    constexpr int KS0 = K0PAD / 32;
    __shared__ _Float16 s_info[4][KS0 * 512];
    __shared__ _Float16 s_h0[4][2 * 512];
    __shared__ _Float16 s_h1[4][2 * 512];

    const int lane = threadIdx.x & 31;
    const int wv   = threadIdx.x >> 5;
    const int e0   = (blockIdx.x * 4 + wv) * 16;
    const int h    = lane >> 4;
    const int col  = lane & 15;

    // ---- stage edge input [x_dst|x_src|ea] into fragment-order LDS -------
    for (int t = lane; t < 16 * (K0PAD / 4); t += 32) {
        int r  = t / (K0PAD / 4);
        int c4 = (t - r * (K0PAD / 4)) * 4;     // k base, multiple of 4
        int eg = e0 + r;
        v4h val = {};
        if (LAYER == 1) {
            float4 f;
            bool have = true;
            if (c4 == 0)      f = *(const float4*)&x4[dstI[eg] * 4];
            else if (c4 == 4) f = *(const float4*)&x4[srcI[eg] * 4];
            else if (c4 == 8) f = *(const float4*)&ea[eg * 4];
            else have = false;
            if (have) { val[0]=(_Float16)f.x; val[1]=(_Float16)f.y;
                        val[2]=(_Float16)f.z; val[3]=(_Float16)f.w; }
        } else {
            if (c4 < 64)        val = *(const v4h*)&h64[dstI[eg] * 64 + c4];
            else if (c4 < 128)  val = *(const v4h*)&h64[srcI[eg] * 64 + (c4 - 64)];
            else if (c4 == 128) {
                float4 f = *(const float4*)&ea[eg * 4];
                val[0]=(_Float16)f.x; val[1]=(_Float16)f.y;
                val[2]=(_Float16)f.z; val[3]=(_Float16)f.w;
            }
        }
        int ks = c4 >> 5, kk = c4 & 31, hh, eb;
        inv_kk(kk, hh, eb);                     // 4-aligned kk -> 4 consecutive e
        *(v4h*)&s_info[wv][ks * 512 + (hh * 16 + r) * 16 + eb] = val;
    }

    const v16h* infF = (const v16h*)s_info[wv];
    const v16h* h0F  = (const v16h*)s_h0[wv];
    const v16h* h1F  = (const v16h*)s_h1[wv];
    const v16h* W0F  = (const v16h*)W0;
    const v16h* W1F  = (const v16h*)W1;
    const v16h* W2F  = (const v16h*)W2;

    // ---- layer 0: [16,K0PAD] @ [K0PAD,64] --------------------------------
    for (int nt = 0; nt < 4; ++nt) {
        v8f acc = {};
        for (int ks = 0; ks < KS0; ++ks) {
            v16h a = infF[ks * 32 + lane];
            v16h b = W0F[(ks * 4 + nt) * 32 + lane];
            acc = __builtin_amdgcn_wmma_f32_16x16x32_f16(false, a, false, b,
                                                         (short)0, acc, false, false);
        }
        int n = nt * 16 + col, hh, e2;
        inv_kk(n & 31, hh, e2);
        float bias = b0[n];
#pragma unroll
        for (int r = 0; r < 8; ++r)
            s_h0[wv][(n >> 5) * 512 + (hh * 16 + r + 8 * h) * 16 + e2] =
                (_Float16)fmaxf(acc[r] + bias, 0.f);
    }

    // ---- layer 1: [16,64] @ [64,64] --------------------------------------
    for (int nt = 0; nt < 4; ++nt) {
        v8f acc = {};
#pragma unroll
        for (int ks = 0; ks < 2; ++ks) {
            v16h a = h0F[ks * 32 + lane];
            v16h b = W1F[(ks * 4 + nt) * 32 + lane];
            acc = __builtin_amdgcn_wmma_f32_16x16x32_f16(false, a, false, b,
                                                         (short)0, acc, false, false);
        }
        int n = nt * 16 + col, hh, e2;
        inv_kk(n & 31, hh, e2);
        float bias = b1[n];
#pragma unroll
        for (int r = 0; r < 8; ++r)
            s_h1[wv][(n >> 5) * 512 + (hh * 16 + r + 8 * h) * 16 + e2] =
                (_Float16)fmaxf(acc[r] + bias, 0.f);
    }

    // ---- layer 2: [16,64] @ [64,256] -> encoded atomic max into agg ------
    unsigned dn[8];
#pragma unroll
    for (int r = 0; r < 8; ++r) dn[r] = (unsigned)dstI[e0 + r + 8 * h] * 256u;
    for (int nt = 0; nt < 16; ++nt) {
        v8f acc = {};
#pragma unroll
        for (int ks = 0; ks < 2; ++ks) {
            v16h a = h1F[ks * 32 + lane];
            v16h b = W2F[(ks * 16 + nt) * 32 + lane];
            acc = __builtin_amdgcn_wmma_f32_16x16x32_f16(false, a, false, b,
                                                         (short)0, acc, false, false);
        }
        int n = nt * 16 + col;
        float bias = b2[n];
#pragma unroll
        for (int r = 0; r < 8; ++r)
            atomicMax(&agg[dn[r] + n], encf(acc[r] + bias));
    }
}

// ------------------------- per-node gamma MLP (small, VALU) ---------------
__global__ __launch_bounds__(64) void gamma_kernel(
    const unsigned* __restrict__ agg, const float* __restrict__ xin, int xdim,
    const float* __restrict__ W0, const float* __restrict__ b0,
    const float* __restrict__ W1, const float* __restrict__ b1,
    const float* __restrict__ W2, const float* __restrict__ b2,
    float* __restrict__ outf, _Float16* __restrict__ outh, int relu_out)
{
    __shared__ float s_in[320];
    __shared__ float s_hid[64];
    __shared__ float s_hid2[64];
    int n = blockIdx.x, tid = threadIdx.x;
    for (int i = tid; i < 256; i += 64) {
        unsigned u = agg[n * 256 + i];
        s_in[i] = (u == ENC_NEGINF) ? 0.f : decf(u);   // empty segments -> 0
    }
    for (int i = tid; i < xdim; i += 64) s_in[256 + i] = xin[n * xdim + i];
    __syncthreads();
    int din = 256 + xdim;
    float acc = b0[tid];
    for (int i = 0; i < din; ++i) acc = fmaf(s_in[i], W0[i * 64 + tid], acc);
    s_hid[tid] = fmaxf(acc, 0.f);
    __syncthreads();
    acc = b1[tid];
    for (int i = 0; i < 64; ++i) acc = fmaf(s_hid[i], W1[i * 64 + tid], acc);
    s_hid2[tid] = fmaxf(acc, 0.f);
    __syncthreads();
    acc = b2[tid];
    for (int i = 0; i < 64; ++i) acc = fmaf(s_hid2[i], W2[i * 64 + tid], acc);
    if (relu_out) acc = fmaxf(acc, 0.f);
    outf[n * 64 + tid] = acc;
    if (outh) outh[n * 64 + tid] = (_Float16)acc;
}

// ------------------------- head: WMMA over K=8256 + gains -----------------
__global__ __launch_bounds__(128) void head_kernel(
    const float* __restrict__ feat,   // [NA,64]
    const float* __restrict__ goal,   // [NA,8192]
    const _Float16* __restrict__ W0h, // frag [258][4][32][16]
    const float* __restrict__ b0,
    const float* __restrict__ W1, const float* __restrict__ b1,  // [64,64]
    const float* __restrict__ W2, const float* __restrict__ b2,  // [64,4]
    const float* __restrict__ x4, const float* __restrict__ goals,
    float* __restrict__ out)          // [NA,2]
{
    __shared__ _Float16 s_a[4][512];  // per-wave A fragment tile (16 rows x 32 k)
    __shared__ float s_h0[4][16 * 64];
    __shared__ float s_h1[4][16 * 64];
    const int lane = threadIdx.x & 31, wv = threadIdx.x >> 5;
    const int n0 = (blockIdx.x * 4 + wv) * 16;
    const int h = lane >> 4, col = lane & 15;
    const v16h* WF = (const v16h*)W0h;

    v8f acc0 = {}, acc1 = {}, acc2 = {}, acc3 = {};
    for (int ks = 0; ks < 258; ++ks) {            // K = 64 (feat) + 8192 (goal)
        // stage 16x32 f32 -> f16 fragments, coalesced float4 loads
#pragma unroll
        for (int t = lane; t < 128; t += 32) {
            int r  = t >> 3;
            int c4 = (t & 7) * 4;
            int kg = ks * 32 + c4;
            const float* sp = (kg < 64) ? &feat[(n0 + r) * 64 + kg]
                                        : &goal[(size_t)(n0 + r) * 8192 + (kg - 64)];
            float4 f = *(const float4*)sp;
            v4h val; val[0]=(_Float16)f.x; val[1]=(_Float16)f.y;
                     val[2]=(_Float16)f.z; val[3]=(_Float16)f.w;
            int hh, eb; inv_kk(c4, hh, eb);
            *(v4h*)&s_a[wv][(hh * 16 + r) * 16 + eb] = val;
        }
        v16h a = ((const v16h*)s_a[wv])[lane];
        acc0 = __builtin_amdgcn_wmma_f32_16x16x32_f16(false, a, false,
                   WF[(ks * 4 + 0) * 32 + lane], (short)0, acc0, false, false);
        acc1 = __builtin_amdgcn_wmma_f32_16x16x32_f16(false, a, false,
                   WF[(ks * 4 + 1) * 32 + lane], (short)0, acc1, false, false);
        acc2 = __builtin_amdgcn_wmma_f32_16x16x32_f16(false, a, false,
                   WF[(ks * 4 + 2) * 32 + lane], (short)0, acc2, false, false);
        acc3 = __builtin_amdgcn_wmma_f32_16x16x32_f16(false, a, false,
                   WF[(ks * 4 + 3) * 32 + lane], (short)0, acc3, false, false);
    }
#pragma unroll
    for (int nt = 0; nt < 4; ++nt) {
        v8f acc = nt == 0 ? acc0 : nt == 1 ? acc1 : nt == 2 ? acc2 : acc3;
        float bias = b0[nt * 16 + col];
#pragma unroll
        for (int r = 0; r < 8; ++r)
            s_h0[wv][(r + 8 * h) * 64 + nt * 16 + col] = fmaxf(acc[r] + bias, 0.f);
    }
    {   // hidden layer 1: lane handles node (lane&15), 32 outputs per half
        int nd = lane & 15, jb = (lane >> 4) * 32;
        for (int j = 0; j < 32; ++j) {
            float a = b1[jb + j];
            for (int i = 0; i < 64; ++i) a = fmaf(s_h0[wv][nd * 64 + i], W1[i * 64 + jb + j], a);
            s_h1[wv][nd * 64 + jb + j] = fmaxf(a, 0.f);
        }
    }
    if (h == 0) {   // final 64->4, gains, control output
        int nd = col, gn = n0 + nd;
        float o[4];
        for (int j = 0; j < 4; ++j) {
            float a = b2[j];
            for (int i = 0; i < 64; ++i) a = fmaf(s_h1[wv][nd * 64 + i], W2[i * 4 + j], a);
            o[j] = 2.f / (1.f + __expf(-a)) + 0.2f;
        }
        float s0 = x4[gn * 4 + 0] - goals[gn * 2 + 0];
        float s1 = x4[gn * 4 + 1] - goals[gn * 2 + 1];
        out[gn * 2 + 0] = -(o[0] * s0 + o[1] * x4[gn * 4 + 2]);
        out[gn * 2 + 1] = -(o[2] * s1 + o[3] * x4[gn * 4 + 3]);
    }
}

// ------------------------- host launch sequence ---------------------------
extern "C" void kernel_launch(void* const* d_in, const int* in_sizes, int n_in,
                              void* d_out, int out_size, void* d_ws, size_t ws_size,
                              hipStream_t stream) {
    const float* x     = (const float*)d_in[0];
    const float* ea    = (const float*)d_in[1];
    const float* goal  = (const float*)d_in[2];
    const float* goals = (const float*)d_in[3];
    const int*   ei    = (const int*)d_in[4];
    const int* srcI = ei, *dstI = ei + NE;
    const float *p1W0=(const float*)d_in[5],  *p1b0=(const float*)d_in[6],
                *p1W1=(const float*)d_in[7],  *p1b1=(const float*)d_in[8],
                *p1W2=(const float*)d_in[9],  *p1b2=(const float*)d_in[10];
    const float *g1W0=(const float*)d_in[11], *g1b0=(const float*)d_in[12],
                *g1W1=(const float*)d_in[13], *g1b1=(const float*)d_in[14],
                *g1W2=(const float*)d_in[15], *g1b2=(const float*)d_in[16];
    const float *p2W0=(const float*)d_in[17], *p2b0=(const float*)d_in[18],
                *p2W1=(const float*)d_in[19], *p2b1=(const float*)d_in[20],
                *p2W2=(const float*)d_in[21], *p2b2=(const float*)d_in[22];
    const float *g2W0=(const float*)d_in[23], *g2b0=(const float*)d_in[24],
                *g2W1=(const float*)d_in[25], *g2b1=(const float*)d_in[26],
                *g2W2=(const float*)d_in[27], *g2b2=(const float*)d_in[28];
    const float *hW0=(const float*)d_in[29],  *hb0=(const float*)d_in[30],
                *hW1=(const float*)d_in[31],  *hb1=(const float*)d_in[32],
                *hW2=(const float*)d_in[33],  *hb2=(const float*)d_in[34];

    char* ws = (char*)d_ws;
    size_t off = 0;
    unsigned* agg = (unsigned*)(ws + off);            off += (size_t)NA * 256 * 4;
    float* h1f    = (float*)(ws + off);               off += (size_t)NA * 64 * 4;
    _Float16* h1h = (_Float16*)(ws + off);            off += (size_t)NA * 64 * 2;
    float* feat   = (float*)(ws + off);               off += (size_t)NA * 64 * 4;
    _Float16* wp1_0 = (_Float16*)(ws + off);          off += 32  * 64 * 2;
    _Float16* wp1_1 = (_Float16*)(ws + off);          off += 64  * 64 * 2;
    _Float16* wp1_2 = (_Float16*)(ws + off);          off += 64  * 256 * 2;
    _Float16* wp2_0 = (_Float16*)(ws + off);          off += 160 * 64 * 2;
    _Float16* wp2_1 = (_Float16*)(ws + off);          off += 64  * 64 * 2;
    _Float16* wp2_2 = (_Float16*)(ws + off);          off += 64  * 256 * 2;
    _Float16* wh0   = (_Float16*)(ws + off);          off += (size_t)8256 * 64 * 2;

    auto cvt = [&](const float* s, _Float16* d, int K, int Kpad, int N) {
        int tot = Kpad * N;
        convert_frag_kernel<<<(tot + 255) / 256, 256, 0, stream>>>(s, d, K, Kpad, N);
    };
    cvt(p1W0, wp1_0, 12, 32, 64);
    cvt(p1W1, wp1_1, 64, 64, 64);
    cvt(p1W2, wp1_2, 64, 64, 256);
    cvt(p2W0, wp2_0, 132, 160, 64);
    cvt(p2W1, wp2_1, 64, 64, 64);
    cvt(p2W2, wp2_2, 64, 64, 256);
    cvt(hW0,  wh0,  8256, 8256, 64);

    // ---- layer 1 ----
    init_agg_kernel<<<NA, 256, 0, stream>>>(agg);
    edge_phi_kernel<1><<<NE / 64, 128, 0, stream>>>(
        x, nullptr, ea, srcI, dstI, wp1_0, p1b0, wp1_1, p1b1, wp1_2, p1b2, agg);
    gamma_kernel<<<NA, 64, 0, stream>>>(agg, x, 4, g1W0, g1b0, g1W1, g1b1, g1W2, g1b2,
                                        h1f, h1h, 1);
    // ---- layer 2 ----
    init_agg_kernel<<<NA, 256, 0, stream>>>(agg);
    edge_phi_kernel<2><<<NE / 64, 128, 0, stream>>>(
        nullptr, h1h, ea, srcI, dstI, wp2_0, p2b0, wp2_1, p2b1, wp2_2, p2b2, agg);
    gamma_kernel<<<NA, 64, 0, stream>>>(agg, h1f, 64, g2W0, g2b0, g2W1, g2b1, g2W2, g2b2,
                                        feat, nullptr, 0);
    // ---- head + gains ----
    head_kernel<<<NA / 64, 128, 0, stream>>>(feat, goal, wh0, hb0, hW1, hb1, hW2, hb2,
                                             x, goals, (float*)d_out);
}
